// Generator_89807766159651
// MI455X (gfx1250) — matrix operand order repaired and checked by reading, not compiled, not run on hardware
//
#include <hip/hip_runtime.h>
#include <hip/hip_bf16.h>

typedef __attribute__((ext_vector_type(16))) __bf16 v16bf;
typedef __attribute__((ext_vector_type(8)))  __bf16 v8bf;
typedef __attribute__((ext_vector_type(8)))  float  v8f;

#define H      350      // hidden size (CONCAT_DIM)
#define KP     352      // H padded to multiple of 32
#define G3     1050     // 3*H gate rows
#define GP     1056     // padded gate rows
#define RZ     700      // 2*H fused r/z rows
#define RZP    704      // padded
#define EMB    50       // output embedding
#define FCP    64       // padded fc rows
#define BATCH  4096
#define TB     32       // batch rows per workgroup
#define TSTEPS 20
#define NWAVE  11
#define NTHREADS (NWAVE * 32)

// bf16 workspace layout (elements)
#define WIH_OFF  0
#define WHH_OFF  (GP * KP)
#define WSUM_OFF (2 * GP * KP)
#define FC_OFF   (2 * GP * KP + RZP * KP)
#define WS_ELEMS (FC_OFF + FCP * KP)

// ---------------- weight prep: fp32 -> zero-padded bf16 (+ fused r/z sum) ----
__global__ void prep_weights(const float* __restrict__ Wih,
                             const float* __restrict__ Whh,
                             const float* __restrict__ fcW,
                             __bf16* __restrict__ ws) {
    int idx = blockIdx.x * blockDim.x + threadIdx.x;
    if (idx >= WS_ELEMS) return;
    float v = 0.0f;
    if (idx < WHH_OFF) {
        int r = idx / KP, k = idx % KP;
        if (r < G3 && k < H) v = Wih[r * H + k];
    } else if (idx < WSUM_OFF) {
        int t = idx - WHH_OFF;
        int r = t / KP, k = t % KP;
        if (r < G3 && k < H) v = Whh[r * H + k];
    } else if (idx < FC_OFF) {
        int t = idx - WSUM_OFF;
        int r = t / KP, k = t % KP;
        if (r < RZ && k < H) v = Wih[r * H + k] + Whh[r * H + k];  // fused r/z
    } else {
        int t = idx - FC_OFF;
        int r = t / KP, k = t % KP;
        if (r < EMB && k < H) v = fcW[r * H + k];
    }
    ws[idx] = (__bf16)v;
}

// ---------------- fragment loaders (documented CDNA5 WMMA layouts) -----------
// A 16x32 bf16: lanes 0-15 -> M=lane, halfs[0:8]=K kc+0..7, halfs[8:16]=kc+16..23
//               lanes 16-31 -> M=lane-16, halfs = kc+8..15 / kc+24..31
__device__ __forceinline__ v16bf load_a_lds(const __bf16* buf, int mrow, int kc, int lane) {
    int klo = kc + ((lane < 16) ? 0 : 8);
    int khi = kc + ((lane < 16) ? 16 : 24);
    v8bf lo = *(const v8bf*)(buf + mrow * KP + klo);
    v8bf hi = *(const v8bf*)(buf + mrow * KP + khi);
    return __builtin_shufflevector(lo, hi, 0,1,2,3,4,5,6,7,8,9,10,11,12,13,14,15);
}

__device__ __forceinline__ v16bf load_a_global_f32(const float* __restrict__ x,
                                                   long grow, int kc, int lane) {
    int klo = kc + ((lane < 16) ? 0 : 8);
    int khi = kc + ((lane < 16) ? 16 : 24);
    const float* p = x + grow * H;
    v16bf a;
#pragma unroll
    for (int i = 0; i < 8; ++i) {
        int k1 = klo + i, k2 = khi + i;
        a[i]     = (__bf16)((k1 < H) ? p[k1] : 0.0f);
        a[8 + i] = (__bf16)((k2 < H) ? p[k2] : 0.0f);
    }
    return a;
}

// B 32x16 bf16: lane%16 = N column; lanes<16 hold K kc+0..7 / kc+16..23, lanes>=16 kc+8..15 / kc+24..31
__device__ __forceinline__ v16bf load_b(const __bf16* __restrict__ Wp, int row,
                                        int kc, int lane) {
    int klo = kc + ((lane < 16) ? 0 : 8);
    int khi = kc + ((lane < 16) ? 16 : 24);
    v8bf lo = *(const v8bf*)(Wp + (long)row * KP + klo);
    v8bf hi = *(const v8bf*)(Wp + (long)row * KP + khi);
    return __builtin_shufflevector(lo, hi, 0,1,2,3,4,5,6,7,8,9,10,11,12,13,14,15);
}

#define WMMA_BF16(A, B, C) \
    __builtin_amdgcn_wmma_f32_16x16x32_bf16(false, (A), false, (B), (short)0, (C), false, false)

// ---------------- fused GRU + projection -------------------------------------
__global__ __launch_bounds__(NTHREADS) void gru_kernel(
    const float* __restrict__ zin, const float* __restrict__ yin,
    const float* __restrict__ start,
    const float* __restrict__ b_ih, const float* __restrict__ b_hh,
    const float* __restrict__ fc_b,
    const __bf16* __restrict__ ws, float* __restrict__ out) {

    __shared__ __align__(32) __bf16 hbuf[2][TB * KP];

    const int tid  = threadIdx.x;
    const int w    = tid >> 5;
    const int lane = tid & 31;
    const int n16  = lane & 15;
    const int rowbase = blockIdx.x * TB;

    const __bf16* Wih  = ws + WIH_OFF;
    const __bf16* Whh  = ws + WHH_OFF;
    const __bf16* Wsum = ws + WSUM_OFF;
    const __bf16* Fc   = ws + FC_OFF;

    const v8f vzero = {0.f, 0.f, 0.f, 0.f, 0.f, 0.f, 0.f, 0.f};

    // ---- init: h0 = concat(z, y) into hbuf[0]; zero pad columns of both bufs
    {
        int c = tid;  // 0..351, exactly KP columns
        for (int m = 0; m < TB; ++m) {
            long gr = rowbase + m;
            float v;
            if (c < 100)     v = zin[gr * 100 + c];
            else if (c < H)  v = yin[gr * 250 + (c - 100)];
            else             v = 0.0f;
            hbuf[0][m * KP + c] = (__bf16)v;
            if (c >= H) hbuf[1][m * KP + c] = (__bf16)0.0f;
        }
    }
    __syncthreads();

    for (int t = 0; t < TSTEPS; ++t) {
        const __bf16* cur = hbuf[t & 1];
        __bf16* nxt = hbuf[(t + 1) & 1];

        const int mrow0 = n16;
        const int mrow1 = 16 + n16;

        // ---- gate GEMMs: each wave handles j-tiles {w, w+11} x both M-tiles.
        // Accumulators per m-tile: [0]=r (i_r+h_r), [1]=z (i_z+h_z), [2]=i_n, [3]=h_n
        for (int js = 0; js < 2; ++js) {
            const int j0 = (w + js * NWAVE) * 16;
            const int bj = j0 + n16;

            v8f acc[2][4];
#pragma unroll
            for (int mi = 0; mi < 2; ++mi)
#pragma unroll
                for (int g = 0; g < 4; ++g) acc[mi][g] = vzero;

            if (t == 0) {
                // pass X: x = start (global fp32), weights W_ih
                for (int kc = 0; kc < KP; kc += 32) {
                    v16bf ax0 = load_a_global_f32(start, rowbase + mrow0, kc, lane);
                    v16bf ax1 = load_a_global_f32(start, rowbase + mrow1, kc, lane);
                    v16bf br = load_b(Wih,         bj, kc, lane);
                    v16bf bz = load_b(Wih,     H + bj, kc, lane);
                    v16bf bn = load_b(Wih, 2 * H + bj, kc, lane);
                    acc[0][0] = WMMA_BF16(ax0, br, acc[0][0]);
                    acc[1][0] = WMMA_BF16(ax1, br, acc[1][0]);
                    acc[0][1] = WMMA_BF16(ax0, bz, acc[0][1]);
                    acc[1][1] = WMMA_BF16(ax1, bz, acc[1][1]);
                    acc[0][2] = WMMA_BF16(ax0, bn, acc[0][2]);
                    acc[1][2] = WMMA_BF16(ax1, bn, acc[1][2]);
                }
                // pass H: h = h0 (LDS), weights W_hh
                for (int kc = 0; kc < KP; kc += 32) {
                    v16bf ah0 = load_a_lds(cur, mrow0, kc, lane);
                    v16bf ah1 = load_a_lds(cur, mrow1, kc, lane);
                    v16bf br = load_b(Whh,         bj, kc, lane);
                    v16bf bz = load_b(Whh,     H + bj, kc, lane);
                    v16bf bn = load_b(Whh, 2 * H + bj, kc, lane);
                    acc[0][0] = WMMA_BF16(ah0, br, acc[0][0]);
                    acc[1][0] = WMMA_BF16(ah1, br, acc[1][0]);
                    acc[0][1] = WMMA_BF16(ah0, bz, acc[0][1]);
                    acc[1][1] = WMMA_BF16(ah1, bz, acc[1][1]);
                    acc[0][3] = WMMA_BF16(ah0, bn, acc[0][3]);
                    acc[1][3] = WMMA_BF16(ah1, bn, acc[1][3]);
                }
            } else {
                // x == h: fused r/z with summed weights; software-pipelined B stream
                v16bf brC  = load_b(Wsum,        bj, 0, lane);
                v16bf bzC  = load_b(Wsum,    H + bj, 0, lane);
                v16bf binC = load_b(Wih, 2 * H + bj, 0, lane);
                v16bf bhnC = load_b(Whh, 2 * H + bj, 0, lane);
#pragma unroll
                for (int kc = 0; kc < KP; kc += 32) {
                    const int kn = (kc + 32 < KP) ? (kc + 32) : 0;  // harmless on last iter
                    v16bf brN  = load_b(Wsum,        bj, kn, lane);
                    v16bf bzN  = load_b(Wsum,    H + bj, kn, lane);
                    v16bf binN = load_b(Wih, 2 * H + bj, kn, lane);
                    v16bf bhnN = load_b(Whh, 2 * H + bj, kn, lane);
                    v16bf ah0 = load_a_lds(cur, mrow0, kc, lane);
                    v16bf ah1 = load_a_lds(cur, mrow1, kc, lane);
                    acc[0][0] = WMMA_BF16(ah0, brC,  acc[0][0]);
                    acc[1][0] = WMMA_BF16(ah1, brC,  acc[1][0]);
                    acc[0][1] = WMMA_BF16(ah0, bzC,  acc[0][1]);
                    acc[1][1] = WMMA_BF16(ah1, bzC,  acc[1][1]);
                    acc[0][2] = WMMA_BF16(ah0, binC, acc[0][2]);
                    acc[1][2] = WMMA_BF16(ah1, binC, acc[1][2]);
                    acc[0][3] = WMMA_BF16(ah0, bhnC, acc[0][3]);
                    acc[1][3] = WMMA_BF16(ah1, bhnC, acc[1][3]);
                    brC = brN; bzC = bzN; binC = binN; bhnC = bhnN;
                }
            }

            // ---- gate nonlinearity + state update (C layout: M=v+8*(lane>=16), N=lane%16)
            const int j = j0 + n16;
            float brz = 0.f, bzz = 0.f, bin_ = 0.f, bhn_ = 0.f;
            if (j < H) {
                brz  = b_ih[j] + b_hh[j];
                bzz  = b_ih[H + j] + b_hh[H + j];
                bin_ = b_ih[2 * H + j];
                bhn_ = b_hh[2 * H + j];
            }
#pragma unroll
            for (int mi = 0; mi < 2; ++mi) {
#pragma unroll
                for (int v = 0; v < 8; ++v) {
                    int m = mi * 16 + v + ((lane >= 16) ? 8 : 0);
                    if (j < H) {
                        float r  = 1.0f / (1.0f + __expf(-(acc[mi][0][v] + brz)));
                        float zg = 1.0f / (1.0f + __expf(-(acc[mi][1][v] + bzz)));
                        float ng = tanhf(acc[mi][2][v] + bin_ + r * (acc[mi][3][v] + bhn_));
                        float hp = (float)cur[m * KP + j];
                        float hv = (1.0f - zg) * ng + zg * hp;
                        nxt[m * KP + j] = (__bf16)hv;
                    }
                }
            }
        }
        __syncthreads();  // h_{t+1} complete; also protects cur for next-step writes

        // ---- fused projection: out[:, t, :] = h_new @ fc_W^T + fc_b
        if (w < 8) {
            const int mi = w >> 2;
            const int e0 = (w & 3) * 16;
            const int mrow = mi * 16 + n16;
            v8f acc = vzero;
            v16bf bC = load_b(Fc, e0 + n16, 0, lane);
#pragma unroll
            for (int kc = 0; kc < KP; kc += 32) {
                const int kn = (kc + 32 < KP) ? (kc + 32) : 0;
                v16bf bN = load_b(Fc, e0 + n16, kn, lane);
                v16bf a = load_a_lds(nxt, mrow, kc, lane);
                acc = WMMA_BF16(a, bC, acc);
                bC = bN;
            }
            const int e = e0 + n16;
            const float fb = (e < EMB) ? fc_b[e] : 0.0f;
#pragma unroll
            for (int v = 0; v < 8; ++v) {
                int m = mi * 16 + v + ((lane >= 16) ? 8 : 0);
                if (e < EMB) {
                    long grow = rowbase + m;
                    out[(grow * TSTEPS + t) * EMB + e] = acc[v] + fb;
                }
            }
        }
        // no barrier needed here: next step writes hbuf[t&1], which nothing reads
        // after the barrier above, and reads hbuf[(t+1)&1], produced before it.
    }
}

// ---------------- host entry -------------------------------------------------
extern "C" void kernel_launch(void* const* d_in, const int* in_sizes, int n_in,
                              void* d_out, int out_size, void* d_ws, size_t ws_size,
                              hipStream_t stream) {
    const float* z     = (const float*)d_in[0];
    const float* y     = (const float*)d_in[1];
    const float* start = (const float*)d_in[2];
    const float* W_ih  = (const float*)d_in[3];
    const float* W_hh  = (const float*)d_in[4];
    const float* b_ih  = (const float*)d_in[5];
    const float* b_hh  = (const float*)d_in[6];
    const float* fc_W  = (const float*)d_in[7];
    const float* fc_b  = (const float*)d_in[8];
    __bf16* ws  = (__bf16*)d_ws;
    float*  out = (float*)d_out;

    (void)in_sizes; (void)n_in; (void)out_size; (void)ws_size;

    const int pt = 256;
    prep_weights<<<(WS_ELEMS + pt - 1) / pt, pt, 0, stream>>>(W_ih, W_hh, fc_W, ws);
    gru_kernel<<<BATCH / TB, NTHREADS, 0, stream>>>(z, y, start, b_ih, b_hh, fc_b, ws, out);
}
